// VeS_33500744909131
// MI455X (gfx1250) — matrix operand order repaired and checked by reading, not compiled, not run on hardware
//
#include <hip/hip_runtime.h>
#include <math.h>
#include <stdint.h>

// ---------------------------------------------------------------------------
// Problem constants (from reference): B=32, Na=Nv=256, D=256
// ---------------------------------------------------------------------------
#define BATCH 32
#define NTOK  256            // Na == Nv
#define DIM   256
#define VSTRIDE (DIM + 8)    // +16B row padding: 16 consecutive rows -> 16 banks

typedef _Float16 v16h __attribute__((ext_vector_type(16)));
typedef _Float16 v8h  __attribute__((ext_vector_type(8)));
typedef float    v8f  __attribute__((ext_vector_type(8)));

#define NEG_INF (-3.4e38f)

// Block-wide sum for 256-thread blocks (8 wave32 waves). `red` is 8 floats of LDS.
__device__ __forceinline__ float blockSum256(float v, float* red) {
    #pragma unroll
    for (int m = 16; m >= 1; m >>= 1) v += __shfl_xor(v, m, 32);
    const int lane = threadIdx.x & 31;
    const int w    = threadIdx.x >> 5;
    __syncthreads();                 // protect red from any previous use
    if (lane == 0) red[w] = v;
    __syncthreads();
    return red[0] + red[1] + red[2] + red[3] + red[4] + red[5] + red[6] + red[7];
}

__device__ __forceinline__ float log_sigmoid(float x) {
    return (x >= 0.0f) ? -log1pf(expf(-x)) : (x - log1pf(expf(x)));
}

// ---------------------------------------------------------------------------
// Kernel 1: L2-normalize each token embedding, convert fp32 -> fp16.
// One 256-thread block per row (D=256). Rows [0, B*Na) = audio, rest visual.
// ---------------------------------------------------------------------------
__global__ __launch_bounds__(256) void ves_norm_kernel(
    const float* __restrict__ audio, const float* __restrict__ visual,
    _Float16* __restrict__ afh, _Float16* __restrict__ vfh)
{
    __shared__ float red[8];
    const int row = blockIdx.x;
    const int t   = threadIdx.x;

    const float* src;
    _Float16*    dst;
    if (row < BATCH * NTOK) {
        src = audio  + (size_t)row * DIM;
        dst = afh    + (size_t)row * DIM;
    } else {
        const int r = row - BATCH * NTOK;
        src = visual + (size_t)r * DIM;
        dst = vfh    + (size_t)r * DIM;
    }
    const float x  = src[t];
    const float ss = blockSum256(x * x, red);
    const float dn = fmaxf(sqrtf(ss), 1e-12f);
    dst[t] = (_Float16)(x / dn);
}

// ---------------------------------------------------------------------------
// Kernel 2: one workgroup per (b,c) pair. S = Af_b (256x256) x Vf_c^T with
// v_wmma_f32_16x16x32_f16, reductions fully fused (S never materialized).
// Each wave holds TWO A tile rows (tm = wave, wave+8) in registers; every
// B fragment read from LDS feeds two WMMAs -> 2x arithmetic intensity per
// LDS byte. V_c is staged with GLOBAL_LOAD_ASYNC_TO_LDS_B128 (ASYNCcnt path,
// no VGPR round-trip). Col-max partials are written once per (wave,half,n):
// no shuffles, no branches, no RMW in the hot loop.
// ---------------------------------------------------------------------------
__global__ __launch_bounds__(256) void ves_pair_kernel(
    const _Float16* __restrict__ afh, const _Float16* __restrict__ vfh,
    const int* __restrict__ attn_mask, const float* __restrict__ logit_scale,
    float* __restrict__ pair_a2v, float* __restrict__ pair_v2a,
    float* __restrict__ pair_nn, float* __restrict__ posmax)
{
    __shared__ _Float16 vbuf[NTOK * VSTRIDE];    // 132 KB (padded vs bank conflicts)
    __shared__ float    rowmax_s[NTOK];          // 1 KB
    __shared__ float    colmax_part[16][NTOK];   // 16 KB: [wave*2 + lane-half][n]
    __shared__ float    nn_part[8];
    __shared__ float    red[8];

    const int c    = blockIdx.x;     // visual batch index
    const int b    = blockIdx.y;     // audio batch index
    const int tid  = threadIdx.x;
    const int lane = tid & 31;
    const int wave = tid >> 5;       // 8 waves
    const int m16  = lane & 15;
    const int kh   = lane >> 4;      // lane-half select (K split per ISA layout)

    // ---- stage V_c into padded LDS via async cache->LDS DMA (16B granules).
    // VDST = per-lane LDS byte address (flat shared addr truncates to LDS
    // offset in [31:0], ISA 10.2), VADDR = 64-bit global address, GV mode.
    {
        const _Float16* src = vfh + (size_t)c * NTOK * DIM;
        for (int i = tid; i < NTOK * (DIM / 8); i += 256) {
            const int row = i >> 5;          // DIM/8 == 32 chunks per row
            const int ch  = i & 31;
            const uint32_t ldsa =
                (uint32_t)(uintptr_t)(vbuf + row * VSTRIDE + ch * 8);
            const uint64_t ga =
                (uint64_t)(uintptr_t)(src + row * DIM + ch * 8);
            asm volatile("global_load_async_to_lds_b128 %0, %1, off"
                         :: "v"(ldsa), "v"(ga) : "memory");
        }
        asm volatile("s_wait_asynccnt 0x0" ::: "memory");
    }
    __syncthreads();

    // ---- A fragments for BOTH owned tile rows, all 8 K-chunks, in registers.
    // ISA 16-bit A layout: lane holds row M=lane%16; halves 0..7 -> K=kh*8..+7,
    // halves 8..15 -> K=16+kh*8..+7 within each K32 chunk.
    const int tm0 = wave, tm1 = wave + 8;
    const _Float16* abase0 = afh + ((size_t)b * NTOK + (size_t)tm0 * 16 + m16) * DIM;
    const _Float16* abase1 = afh + ((size_t)b * NTOK + (size_t)tm1 * 16 + m16) * DIM;
    v16h afr0[8], afr1[8];
    #pragma unroll
    for (int kc = 0; kc < 8; ++kc) {
        const v8h l0 = *(const v8h*)(abase0 + kc * 32 + kh * 8);
        const v8h h0 = *(const v8h*)(abase0 + kc * 32 + kh * 8 + 16);
        afr0[kc] = __builtin_shufflevector(l0, h0,
            0, 1, 2, 3, 4, 5, 6, 7, 8, 9, 10, 11, 12, 13, 14, 15);
        const v8h l1 = *(const v8h*)(abase1 + kc * 32 + kh * 8);
        const v8h h1 = *(const v8h*)(abase1 + kc * 32 + kh * 8 + 16);
        afr1[kc] = __builtin_shufflevector(l1, h1,
            0, 1, 2, 3, 4, 5, 6, 7, 8, 9, 10, 11, 12, 13, 14, 15);
    }

    float rm0[8], rm1[8];
    #pragma unroll
    for (int j = 0; j < 8; ++j) { rm0[j] = NEG_INF; rm1[j] = NEG_INF; }
    float nn_acc = 0.0f;

    for (int tn = 0; tn < 16; ++tn) {
        v8f acc0 = {}, acc1 = {};
        #pragma unroll
        for (int kc = 0; kc < 8; ++kc) {
            // B fragment: lane holds column N=lane%16 (a V row); K range by half.
            const _Float16* bbase =
                vbuf + ((size_t)tn * 16 + m16) * VSTRIDE + kc * 32 + kh * 8;
            const v8h lo = *(const v8h*)bbase;
            const v8h hi = *(const v8h*)(bbase + 16);
            const v16h bfr = __builtin_shufflevector(lo, hi,
                0, 1, 2, 3, 4, 5, 6, 7, 8, 9, 10, 11, 12, 13, 14, 15);
            acc0 = __builtin_amdgcn_wmma_f32_16x16x32_f16(
                false, afr0[kc], false, bfr, (short)0, acc0, false, false);
            acc1 = __builtin_amdgcn_wmma_f32_16x16x32_f16(
                false, afr1[kc], false, bfr, (short)0, acc1, false, false);
        }
        // C layout: vgpr j, lane L -> (M = tm*16 + j + 8*kh, N = tn*16 + m16)
        float cm = NEG_INF;
        #pragma unroll
        for (int j = 0; j < 8; ++j) {
            const float s0 = acc0[j], s1 = acc1[j];
            const float g0 = fminf(s0, 0.0f), g1 = fminf(s1, 0.0f);
            nn_acc += g0 * g0 + g1 * g1;        // relu(-s)^2
            rm0[j] = fmaxf(rm0[j], s0);
            rm1[j] = fmaxf(rm1[j], s1);
            cm = fmaxf(cm, fmaxf(s0, s1));
        }
        // one unconditional store per lane: partial col-max for this (half, n)
        colmax_part[wave * 2 + kh][tn * 16 + m16] = cm;
    }

    // ---- row max: reduce across the 16 N-lanes within each lane-half ----
    #pragma unroll
    for (int j = 0; j < 8; ++j) {
        float r0 = rm0[j], r1 = rm1[j];
        r0 = fmaxf(r0, __shfl_xor(r0, 1, 32));
        r0 = fmaxf(r0, __shfl_xor(r0, 2, 32));
        r0 = fmaxf(r0, __shfl_xor(r0, 4, 32));
        r0 = fmaxf(r0, __shfl_xor(r0, 8, 32));
        r1 = fmaxf(r1, __shfl_xor(r1, 1, 32));
        r1 = fmaxf(r1, __shfl_xor(r1, 2, 32));
        r1 = fmaxf(r1, __shfl_xor(r1, 4, 32));
        r1 = fmaxf(r1, __shfl_xor(r1, 8, 32));
        if (m16 == 0) {
            rowmax_s[tm0 * 16 + kh * 8 + j] = r0;
            rowmax_s[tm1 * 16 + kh * 8 + j] = r1;
        }
    }

    // nonneg partial per wave
    #pragma unroll
    for (int m = 16; m >= 1; m >>= 1) nn_acc += __shfl_xor(nn_acc, m, 32);
    if (lane == 0) nn_part[wave] = nn_acc;
    __syncthreads();

    // ---- final per-pair reduction (thread t handles token index t) ----
    const int t = tid;
    float cmax = colmax_part[0][t];
    #pragma unroll
    for (int w = 1; w < 16; ++w) cmax = fmaxf(cmax, colmax_part[w][t]);
    const float rmax = rowmax_s[t];
    const float mval = (float)attn_mask[b * NTOK + t];

    const float v2a_sum = blockSum256(cmax, red);
    const float a2v_num = blockSum256(rmax * mval, red);
    const float a2v_den = blockSum256(mval, red);
    const float nn_tot  = blockSum256((t < 8) ? nn_part[t] : 0.0f, red);

    const float scale = expf(logit_scale[0]);
    if (t == 0) {
        const int p = b * BATCH + c;
        pair_a2v[p] = scale * a2v_num / fmaxf(a2v_den, 1e-5f);
        pair_v2a[p] = scale * v2a_sum * (1.0f / (float)NTOK);
        pair_nn[p]  = scale * scale * nn_tot;
    }
    if (b == c) posmax[b * NTOK + t] = scale * rmax;
}

// ---------------------------------------------------------------------------
// Kernel 3: single block. SigLIP loss + nonneg mean + TV smoothing -> d_out[0].
// ---------------------------------------------------------------------------
__global__ __launch_bounds__(256) void ves_final_kernel(
    const float* __restrict__ pair_a2v, const float* __restrict__ pair_v2a,
    const float* __restrict__ pair_nn, const float* __restrict__ posmax,
    const int* __restrict__ attn_mask, const float* __restrict__ bias,
    float* __restrict__ out)
{
    __shared__ float red[8];
    const int t = threadIdx.x;
    const float b0 = bias[0];

    float lc = 0.0f, nn = 0.0f, tvn = 0.0f, tvd = 0.0f;
    for (int p = t; p < BATCH * BATCH; p += 256) {
        const int bb = p >> 5, cc = p & 31;
        const float cs  = 0.5f * (pair_a2v[p] + pair_v2a[p]);
        const float lab = (bb == cc) ? 1.0f : -1.0f;
        lc -= log_sigmoid(lab * (cs + b0));
        nn += pair_nn[p];
    }
    for (int i = t; i < BATCH * (NTOK - 1); i += 256) {
        const int bb = i / (NTOK - 1);
        const int a  = i - bb * (NTOK - 1);
        const float d  = posmax[bb * NTOK + a + 1] - posmax[bb * NTOK + a];
        const float pm = (float)(attn_mask[bb * NTOK + a + 1] *
                                 attn_mask[bb * NTOK + a]);
        tvn += d * d * pm;
        tvd += pm;
    }
    const float lc_s  = blockSum256(lc, red);
    const float nn_s  = blockSum256(nn, red);
    const float tvn_s = blockSum256(tvn, red);
    const float tvd_s = blockSum256(tvd, red);
    if (t == 0) {
        const float loss_c    = lc_s / (float)(BATCH * BATCH);
        const float l_nonneg  = nn_s /
            ((float)(BATCH * BATCH) * (float)NTOK * (float)NTOK);
        const float l_tv      = tvn_s / fmaxf(tvd_s, 1.0f);
        out[0] = loss_c + l_nonneg + 1e-4f * l_tv;
    }
}

// ---------------------------------------------------------------------------
// Host launcher
// ---------------------------------------------------------------------------
extern "C" void kernel_launch(void* const* d_in, const int* in_sizes, int n_in,
                              void* d_out, int out_size, void* d_ws, size_t ws_size,
                              hipStream_t stream) {
    (void)in_sizes; (void)n_in; (void)out_size; (void)ws_size;
    const float* audio       = (const float*)d_in[0];   // (32,256,256) f32
    const float* visual      = (const float*)d_in[1];   // (32,256,256) f32
    const int*   attn_mask   = (const int*)d_in[2];     // (32,256) i32
    const float* logit_scale = (const float*)d_in[3];   // (1,)
    const float* bias        = (const float*)d_in[4];   // (1,)
    float*       out         = (float*)d_out;           // scalar loss

    // Workspace layout (~8.4 MB): fp16 normalized features + partial outputs.
    const size_t NFEAT = (size_t)BATCH * NTOK * DIM;    // 2,097,152 elems
    _Float16* afh = (_Float16*)d_ws;
    _Float16* vfh = afh + NFEAT;
    float* fbase  = (float*)(vfh + NFEAT);
    float* pair_a2v = fbase;                            // 1024
    float* pair_v2a = fbase + 1024;                     // 1024
    float* pair_nn  = fbase + 2048;                     // 1024
    float* posmax   = fbase + 3072;                     // 32*256 = 8192

    // 1) normalize + fp16 convert: one block per token row
    ves_norm_kernel<<<2 * BATCH * NTOK, 256, 0, stream>>>(audio, visual, afh, vfh);

    // 2) fused WMMA GEMM + reductions: one workgroup per (b,c) pair
    dim3 grid(BATCH, BATCH);
    ves_pair_kernel<<<grid, 256, 0, stream>>>(afh, vfh, attn_mask, logit_scale,
                                              pair_a2v, pair_v2a, pair_nn, posmax);

    // 3) scalar loss assembly
    ves_final_kernel<<<1, 256, 0, stream>>>(pair_a2v, pair_v2a, pair_nn, posmax,
                                            attn_mask, bias, out);
}
